// MLPExtractor_64037962383520
// MI455X (gfx1250) — compile-verified, matrix operand order checked by reading, not run on hardware
//
#include <hip/hip_runtime.h>

#define BATCH  256
#define NN     100
#define MAXN   120
#define HDIM   64
#define KPAIRS 512
#define EFROW  164   // H + N_NODES
#define EFROWS 101

typedef __attribute__((ext_vector_type(16))) _Float16 v16h;
typedef __attribute__((ext_vector_type(8)))  _Float16 v8h;
typedef __attribute__((ext_vector_type(8)))  float    v8f;

// LDS row strides (halves). +8 halves (16B) padding: keeps 16B alignment,
// spreads banks (stride%64 dwords != 0).
#define XS  136   // X chunk: 128 rows x 128 cols
#define WS1 136   // W1t: 64 rows x 128 cols
#define WS2 72    // W2t/W3t: 64 x 64
#define HS  72    // H1/H2: 128 x 64

// Build a 16x32 f16 WMMA A/B fragment. `p` points at the lane's row start +
// kbase + koff. Per ISA layout, halves 0..7 = K koff..koff+7 and halves
// 8..15 = K koff+16..koff+23 -> two contiguous 16B LDS loads.
__device__ __forceinline__ v16h frag_ld(const _Float16* p) {
  v8h lo = *(const v8h*)(p);
  v8h hi = *(const v8h*)(p + 16);
  return __builtin_shufflevector(lo, hi, 0,1,2,3,4,5,6,7,8,9,10,11,12,13,14,15);
}

// ---------------- kernel 1: ordered first-K-nonzero compaction -------------
__global__ void compact_kernel(const float* __restrict__ ef, int* __restrict__ idx) {
  const int b = blockIdx.x;
  const int lane = threadIdx.x;              // 32 threads = 1 wave
  const float* base = ef + (size_t)b * EFROWS * EFROW;
  for (int i = lane; i < KPAIRS; i += 32) idx[b * KPAIRS + i] = 0;  // fill_value=0
  int cnt = 0;
  for (int off = 0; off < NN * NN && cnt < KPAIRS; off += 32) {
    int m = off + lane;
    bool pred = false;
    if (m < NN * NN) {
      int r = m / NN, c = m % NN;
      pred = base[(size_t)(1 + r) * EFROW + HDIM + c] > 0.5f;
    }
    unsigned bal = __builtin_amdgcn_ballot_w32(pred);
    int prefix = __popc(bal & ((1u << lane) - 1u));
    int p = cnt + prefix;
    if (pred && p < KPAIRS) idx[b * KPAIRS + p] = m;
    cnt += __popc(bal);
  }
}

// ---------------- kernel 2: critic MLP (tiny, plain VALU) ------------------
__global__ void critic_kernel(const float* __restrict__ ef,
                              const float* __restrict__ W1, const float* __restrict__ b1,
                              const float* __restrict__ W2, const float* __restrict__ b2,
                              float* __restrict__ out_value) {
  const int b = blockIdx.x;
  const int t = threadIdx.x;                 // 64 threads
  __shared__ float g[64];
  __shared__ float h[64];
  g[t] = ef[(size_t)b * EFROWS * EFROW + t]; // graph row, first 64 features
  __syncthreads();
  float acc = b1[t];
  for (int k = 0; k < 64; ++k) acc = fmaf(g[k], W1[k * 64 + t], acc);
  h[t] = fmaxf(acc, 0.f) * W2[t];            // fold W2 (64,1) into reduction
  __syncthreads();
  for (int s = 32; s > 0; s >>= 1) { if (t < s) h[t] += h[t + s]; __syncthreads(); }
  if (t == 0) out_value[b] = h[0] + b2[0];
}

// ---------------- kernel 3: fused actor: gather + 3 WMMA GEMMs + dot +
// softmax + scatter, one workgroup (8 waves) per batch ----------------------
__global__ __launch_bounds__(256)
void actor_kernel(const float* __restrict__ ef,
                  const float* __restrict__ W1, const float* __restrict__ W2,
                  const float* __restrict__ W3, const float* __restrict__ W4,
                  const float* __restrict__ b1, const float* __restrict__ b2,
                  const float* __restrict__ b3, const float* __restrict__ b4,
                  const int* __restrict__ idxg, float* __restrict__ out) {
  __shared__ __align__(16) _Float16 sX[128 * XS];
  __shared__ __align__(16) _Float16 sW1t[64 * WS1];
  __shared__ __align__(16) _Float16 sW2t[64 * WS2];
  __shared__ __align__(16) _Float16 sW3t[64 * WS2];
  __shared__ __align__(16) _Float16 sH1[128 * HS];
  __shared__ __align__(16) _Float16 sH2[128 * HS];
  __shared__ float sg[64], g1[64], sb2[64], sb3[64], sw4[64];
  __shared__ float logits[KPAIRS];
  __shared__ float red[256];
  __shared__ int sidx[KPAIRS];

  const int t = threadIdx.x;
  const int b = blockIdx.x;
  const float* efb = ef + (size_t)b * EFROWS * EFROW;
  float* orow = out + (size_t)b * (MAXN * MAXN);

  // zero this batch's padded 120x120 output plane
  for (int i = t; i < MAXN * MAXN; i += 256) orow[i] = 0.f;

  if (t < 64) { sg[t] = efb[t]; sb2[t] = b2[t]; sb3[t] = b3[t]; sw4[t] = W4[t]; }
  for (int i = t; i < KPAIRS; i += 256) sidx[i] = idxg[b * KPAIRS + i];
  __syncthreads();

  // graph @ W1[0:64] + b1 -> per-batch constant row folded into layer-1 bias
  if (t < 64) {
    float a = b1[t];
    for (int k = 0; k < 64; ++k) a = fmaf(sg[k], W1[k * 64 + t], a);
    g1[t] = a;
  }
  // weights -> f16, transposed [n][k] so B-fragments are k-contiguous
  for (int i = t; i < 64 * 128; i += 256) {
    int n = i & 63, k = i >> 6;
    sW1t[n * WS1 + k] = (_Float16)W1[(64 + k) * 64 + n];   // rows 64..191 (n1|n2 part)
  }
  for (int i = t; i < 64 * 64; i += 256) {
    int n = i & 63, k = i >> 6;
    sW2t[n * WS2 + k] = (_Float16)W2[k * 64 + n];
    sW3t[n * WS2 + k] = (_Float16)W3[k * 64 + n];
  }
  const float b4s = b4[0];
  __syncthreads();

  const int wv   = t >> 5;            // wave id 0..7 -> M-tile
  const int lane = t & 31;
  const int ln   = lane & 15;
  const int koff = (lane >> 4) * 8;
  const int arow = wv * 16 + ln;      // A-fragment row within 128-row chunk
  const int mbase = wv * 16 + (lane >> 4) * 8;  // C/D row base for this lane

  for (int ch = 0; ch < 4; ++ch) {
    // ---- gather 128 pair rows: [n1 | n2] as f16, 2 threads per row --------
    {
      int rr = t >> 1, half = t & 1;
      int v = sidx[ch * 128 + rr];
      int node = half ? (v / NN) : (v % NN);   // i2 = idx//n, i1 = idx%n
      const float4* src = (const float4*)(efb + (size_t)(1 + node) * EFROW);
      _Float16* dst = &sX[rr * XS + half * 64];
      for (int k = 0; k < 16; ++k) {
        float4 f = src[k];
        dst[4 * k + 0] = (_Float16)f.x; dst[4 * k + 1] = (_Float16)f.y;
        dst[4 * k + 2] = (_Float16)f.z; dst[4 * k + 3] = (_Float16)f.w;
      }
    }
    __syncthreads();

    v8f acc[4];
    // ---- layer 1: [128x128] @ [128x64], bias = g1 (states part folded) ----
    for (int nt = 0; nt < 4; ++nt) {
      float gv = g1[nt * 16 + ln];
      for (int j = 0; j < 8; ++j) acc[nt][j] = gv;
    }
    for (int kb = 0; kb < 128; kb += 32) {
      v16h af = frag_ld(&sX[arow * XS + kb + koff]);
      for (int nt = 0; nt < 4; ++nt) {
        v16h bf = frag_ld(&sW1t[(nt * 16 + ln) * WS1 + kb + koff]);
        acc[nt] = __builtin_amdgcn_wmma_f32_16x16x32_f16(false, af, false, bf,
                                                         (short)0, acc[nt], false, false);
      }
    }
    for (int nt = 0; nt < 4; ++nt)
      for (int j = 0; j < 8; ++j)
        sH1[(mbase + j) * HS + nt * 16 + ln] = (_Float16)fmaxf(acc[nt][j], 0.f);
    __syncthreads();

    // ---- layer 2: [128x64] @ [64x64] --------------------------------------
    for (int nt = 0; nt < 4; ++nt) {
      float bv = sb2[nt * 16 + ln];
      for (int j = 0; j < 8; ++j) acc[nt][j] = bv;
    }
    for (int kb = 0; kb < 64; kb += 32) {
      v16h af = frag_ld(&sH1[arow * HS + kb + koff]);
      for (int nt = 0; nt < 4; ++nt) {
        v16h bf = frag_ld(&sW2t[(nt * 16 + ln) * WS2 + kb + koff]);
        acc[nt] = __builtin_amdgcn_wmma_f32_16x16x32_f16(false, af, false, bf,
                                                         (short)0, acc[nt], false, false);
      }
    }
    for (int nt = 0; nt < 4; ++nt)
      for (int j = 0; j < 8; ++j)
        sH2[(mbase + j) * HS + nt * 16 + ln] = (_Float16)fmaxf(acc[nt][j], 0.f);
    __syncthreads();

    // ---- layer 3 + fused layer 4 (64->1 dot) ------------------------------
    for (int nt = 0; nt < 4; ++nt) {
      float bv = sb3[nt * 16 + ln];
      for (int j = 0; j < 8; ++j) acc[nt][j] = bv;
    }
    for (int kb = 0; kb < 64; kb += 32) {
      v16h af = frag_ld(&sH2[arow * HS + kb + koff]);
      for (int nt = 0; nt < 4; ++nt) {
        v16h bf = frag_ld(&sW3t[(nt * 16 + ln) * WS2 + kb + koff]);
        acc[nt] = __builtin_amdgcn_wmma_f32_16x16x32_f16(false, af, false, bf,
                                                         (short)0, acc[nt], false, false);
      }
    }
    {
      float pl[8];
      for (int j = 0; j < 8; ++j) pl[j] = 0.f;
      for (int nt = 0; nt < 4; ++nt) {
        float w4v = sw4[nt * 16 + ln];
        for (int j = 0; j < 8; ++j) pl[j] = fmaf(fmaxf(acc[nt][j], 0.f), w4v, pl[j]);
      }
      // lanes 0..15 all hold row mbase+j; lanes 16..31 hold row mbase+j (own mbase)
      for (int j = 0; j < 8; ++j) {
        float s = pl[j];
        s += __shfl_xor(s, 1);
        s += __shfl_xor(s, 2);
        s += __shfl_xor(s, 4);
        s += __shfl_xor(s, 8);
        if (ln == 0) logits[ch * 128 + mbase + j] = s + b4s;
      }
    }
    __syncthreads();
  }

  // ---- softmax over 512 logits ---------------------------------------------
  float l0 = logits[t], l1 = logits[t + 256];
  red[t] = fmaxf(l0, l1);
  __syncthreads();
  for (int s = 128; s > 0; s >>= 1) { if (t < s) red[t] = fmaxf(red[t], red[t + s]); __syncthreads(); }
  float mx = red[0];
  __syncthreads();
  float e0 = __expf(l0 - mx), e1 = __expf(l1 - mx);
  red[t] = e0 + e1;
  __syncthreads();
  for (int s = 128; s > 0; s >>= 1) { if (t < s) red[t] += red[t + s]; __syncthreads(); }
  float inv = 1.f / red[0];

  // ---- scatter pi into padded 120x120 plane --------------------------------
  {
    int v0 = sidx[t], v1 = sidx[t + 256];
    orow[(v0 / NN) * MAXN + (v0 % NN)] = e0 * inv;
    orow[(v1 / NN) * MAXN + (v1 % NN)] = e1 * inv;
  }
}

extern "C" void kernel_launch(void* const* d_in, const int* in_sizes, int n_in,
                              void* d_out, int out_size, void* d_ws, size_t ws_size,
                              hipStream_t stream) {
  const float* ef  = (const float*)d_in[0];
  const float* aW0 = (const float*)d_in[1];   // (192,64)
  const float* aW1 = (const float*)d_in[2];   // (64,64)
  const float* aW2 = (const float*)d_in[3];   // (64,64)
  const float* aW3 = (const float*)d_in[4];   // (64,1)
  const float* ab0 = (const float*)d_in[5];
  const float* ab1 = (const float*)d_in[6];
  const float* ab2 = (const float*)d_in[7];
  const float* ab3 = (const float*)d_in[8];
  const float* cW0 = (const float*)d_in[9];   // (64,64)
  const float* cW1 = (const float*)d_in[10];  // (64,1)
  const float* cb0 = (const float*)d_in[11];
  const float* cb1 = (const float*)d_in[12];

  int*   idx = (int*)d_ws;                    // BATCH*KPAIRS ints
  float* out = (float*)d_out;                 // filled (B*14400) then value (B)
  float* out_value = out + (size_t)BATCH * MAXN * MAXN;

  compact_kernel<<<BATCH, 32, 0, stream>>>(ef, idx);
  critic_kernel<<<BATCH, 64, 0, stream>>>(ef, cW0, cb0, cW1, cb1, out_value);
  actor_kernel<<<BATCH, 256, 0, stream>>>(ef, aW0, aW1, aW2, aW3,
                                          ab0, ab1, ab2, ab3, idx, out);
}
